// Glm4MoE_36739150250370
// MI455X (gfx1250) — matrix operand order repaired and checked by reading, not compiled, not run on hardware
//
#include <hip/hip_runtime.h>
#include <hip/hip_bf16.h>

// ---------------- problem constants ----------------
#define NUM_EXPERTS 32
#define TOPK 8
#define HID 2048
#define INTER 1408
#define NTOK 2048
#define NASSIGN (NTOK * TOPK)          // 16384 routed rows
#define NROWS (NASSIGN + NTOK)         // + shared-expert rows = 18432
#define NGROUPS (NUM_EXPERTS + 1)      // expert 32 == shared expert

// ---------------- tile config ----------------
#define TM 128    // rows per block (more rows => fewer weight re-reads)
#define TN 64     // cols per block
#define TKK 32    // k per stage
#define LDAP 36   // padded LDS stride (bf16): 18 words, conflict-free, 8B-aligned rows
#define LDAA 40   // mlp2 A stride (bf16): 20 words, conflict-free, 16B-aligned rows

typedef __attribute__((ext_vector_type(16))) __bf16 v16bf;
typedef __attribute__((ext_vector_type(8)))  float  v8f;
typedef __attribute__((ext_vector_type(4)))  __bf16 bf16x4;
typedef __attribute__((ext_vector_type(2)))  __bf16 bf16x2;

// GCC-style vector matching the async-LDS builtin's parameter type
typedef int v4i_g __attribute__((vector_size(16)));
typedef __attribute__((address_space(1))) v4i_g* glob_v4i_p;
typedef __attribute__((address_space(3))) v4i_g* lds_v4i_p;

// ---- feature probes --------------------------------------------------
#if defined(__has_builtin)
#  if __has_builtin(__builtin_amdgcn_global_load_async_to_lds_b128)
#    define ASYNC_LDS 1
#  endif
#  if __has_builtin(__builtin_amdgcn_s_wait_asynccnt)
#    define HAVE_WAIT_ASYNC 1
#  endif
#  if __has_builtin(__builtin_amdgcn_cvt_pk_bf16_f32)
#    define HAVE_PK_BF16 1
#  endif
#endif

__device__ __forceinline__ bf16x2 f2bf2(float a, float b) {
#ifdef HAVE_PK_BF16
    // single v_cvt_pk_bf16_f32 (RNE); bit_cast tolerates any 4-byte return type
    return __builtin_bit_cast(bf16x2, __builtin_amdgcn_cvt_pk_bf16_f32(a, b));
#else
    bf16x2 r;
    r[0] = (__bf16)a;
    r[1] = (__bf16)b;
    return r;
#endif
}

__device__ __forceinline__ void async_copy16(const void* g, void* l) {
#ifdef ASYNC_LDS
    __builtin_amdgcn_global_load_async_to_lds_b128(
        (glob_v4i_p)(void*)const_cast<void*>(g),
        (lds_v4i_p)l, 0, 0);
#else
    *(uint4*)l = *(const uint4*)g;
#endif
}

__device__ __forceinline__ void wait_async() {
#ifdef ASYNC_LDS
#  ifdef HAVE_WAIT_ASYNC
    __builtin_amdgcn_s_wait_asynccnt(0);
#  else
    asm volatile("s_wait_asynccnt 0" ::: "memory");
#  endif
#endif
}

__device__ __forceinline__ v8f wmma_bf16(v16bf a, v16bf b, v8f c) {
    return __builtin_amdgcn_wmma_f32_16x16x32_bf16(
        false, a, false, b, (short)0, c, false, false);
}

// =====================================================================
// 1) Gating: one wave per token; lane e computes logit for expert e.
// =====================================================================
__global__ __launch_bounds__(32) void gate_kernel(
    const float* __restrict__ x, const float* __restrict__ gw,
    const float* __restrict__ gb, int* __restrict__ topk,
    float* __restrict__ wts)
{
    int t = blockIdx.x;
    int e = threadIdx.x;
    const float4* xr = (const float4*)(x + (size_t)t * HID);
    const float4* wr = (const float4*)(gw + (size_t)e * HID);
    float acc = 0.f;
    for (int i = 0; i < HID / 4; ++i) {
        float4 a = xr[i], b = wr[i];
        acc += a.x * b.x + a.y * b.y + a.z * b.z + a.w * b.w;
    }
    float s = 1.f / (1.f + expf(-acc));
    __shared__ float sc[32], sb[32];
    sc[e] = s;
    sb[e] = s + gb[e];
    __syncthreads();
    if (e == 0) {
        float wsum = 0.f;
        int idx[TOPK];
        float val[TOPK];
        for (int it = 0; it < TOPK; ++it) {
            float best = -1e30f;
            int be = 0;
            for (int j = 0; j < 32; ++j)
                if (sb[j] > best) { best = sb[j]; be = j; }
            idx[it] = be;
            val[it] = sc[be];
            sb[be] = -1e30f;
            wsum += sc[be];
        }
        float inv = 1.f / wsum;
        for (int it = 0; it < TOPK; ++it) {
            topk[t * TOPK + it] = idx[it];
            wts[t * TOPK + it]  = val[it] * inv;   // ROUTED_SCALING == 1
        }
    }
}

// =====================================================================
// 2-5) dispatch bookkeeping
// =====================================================================
__global__ void zero_counts_kernel(int* __restrict__ counts) {
    if (threadIdx.x < NUM_EXPERTS) counts[threadIdx.x] = 0;
}

__global__ void hist_kernel(const int* __restrict__ topk, int* __restrict__ counts) {
    int i = blockIdx.x * 256 + threadIdx.x;
    if (i < NASSIGN) atomicAdd(&counts[topk[i]], 1);
}

__global__ void scan_kernel(const int* __restrict__ counts,
                            int* __restrict__ offsets, int* __restrict__ cursor) {
    if (threadIdx.x == 0) {
        int acc = 0;
        for (int e = 0; e < NUM_EXPERTS; ++e) {
            offsets[e] = acc;
            cursor[e]  = acc;
            acc += counts[e];
        }
        offsets[NUM_EXPERTS]     = NASSIGN;
        offsets[NUM_EXPERTS + 1] = NROWS;
    }
}

__global__ void scatter_kernel(const int* __restrict__ topk, int* __restrict__ cursor,
                               int* __restrict__ row_token, int* __restrict__ row_slot) {
    int i = blockIdx.x * 256 + threadIdx.x;
    if (i < NASSIGN) {
        int e   = topk[i];
        int pos = atomicAdd(&cursor[e], 1);
        row_token[pos] = i >> 3;
        row_slot[pos]  = i;
    } else if (i < NROWS) {
        row_token[i] = i - NASSIGN;
        row_slot[i]  = i;
    }
}

// =====================================================================
// 6) mlp1: h = silu(x @ w1[e]) * (x @ w3[e])   (grouped, bf16 WMMA)
//    grid: (NTOK/TM=16, INTER/TN=22, NGROUPS=33), block 256 (8 waves)
//    wave tile: 32 rows x 32 cols  (2 A-frags x 2 B-frags, x2 matrices)
// =====================================================================
__global__ __launch_bounds__(256) void mlp1_kernel(
    const float* __restrict__ x,
    const float* __restrict__ w1, const float* __restrict__ w3,
    const float* __restrict__ sw1, const float* __restrict__ sw3,
    const int* __restrict__ row_token, const int* __restrict__ offsets,
    __bf16* __restrict__ hbuf)
{
    int e = blockIdx.z;
    int gStart = offsets[e], gEnd = offsets[e + 1];
    int rowBase = gStart + blockIdx.x * TM;
    if (rowBase >= gEnd) return;
    int i0 = blockIdx.y * TN;

    const float* B1 = (e < NUM_EXPERTS) ? (w1 + (size_t)e * HID * INTER) : sw1;
    const float* B3 = (e < NUM_EXPERTS) ? (w3 + (size_t)e * HID * INTER) : sw3;

    __shared__ __bf16 sA[TM * LDAP];
    __shared__ __bf16 sB1[TN * LDAP];
    __shared__ __bf16 sB3[TN * LDAP];
    __shared__ int sTok[TM];

    int tid = threadIdx.x;
    int lane = tid & 31, wv = tid >> 5;
    int wM = wv & 3, wN = wv >> 2;       // 4 waves x 32 rows, 2 waves x 32 cols
    int l15 = lane & 15;
    int hiHalf = (lane >= 16);

    if (tid < TM) {
        int p = rowBase + tid;
        sTok[tid] = (p < gEnd) ? row_token[p] : row_token[gStart];
    }

    v8f accG[2][2] = {};
    v8f accU[2][2] = {};

    for (int k0 = 0; k0 < HID; k0 += TKK) {
        __syncthreads();
        // ---- stage A: 128 rows x 32 k, float4 loads, bf16x4 LDS stores
        {
            int kk4 = (tid & 7) * 4;
            int rb  = tid >> 3;
            #pragma unroll
            for (int it = 0; it < 4; ++it) {
                int r = rb + it * 32;
                int tok = sTok[r];
                float4 v = *(const float4*)&x[(size_t)tok * HID + k0 + kk4];
                bf16x2 p0 = f2bf2(v.x, v.y), p1 = f2bf2(v.z, v.w);
                bf16x4 pk;
                pk[0] = p0[0]; pk[1] = p0[1]; pk[2] = p1[0]; pk[3] = p1[1];
                *(bf16x4*)&sA[r * LDAP + kk4] = pk;
            }
        }
        // ---- stage B transposed to [col][k], float4 loads along cols
        {
            int ii4 = (tid & 15) * 4;
            int kq  = tid >> 4;
            #pragma unroll
            for (int it = 0; it < 2; ++it) {
                int kk = kq + it * 16;
                size_t gi = (size_t)(k0 + kk) * INTER + i0 + ii4;
                float4 a1 = *(const float4*)&B1[gi];
                float4 a3 = *(const float4*)&B3[gi];
                if (k0 + TKK < HID) {   // prefetch next k-tile rows
                    __builtin_prefetch(&B1[gi + (size_t)TKK * INTER], 0, 1);
                    __builtin_prefetch(&B3[gi + (size_t)TKK * INTER], 0, 1);
                }
                bf16x2 p10 = f2bf2(a1.x, a1.y), p11 = f2bf2(a1.z, a1.w);
                bf16x2 p30 = f2bf2(a3.x, a3.y), p31 = f2bf2(a3.z, a3.w);
                sB1[(ii4 + 0) * LDAP + kk] = p10[0];
                sB1[(ii4 + 1) * LDAP + kk] = p10[1];
                sB1[(ii4 + 2) * LDAP + kk] = p11[0];
                sB1[(ii4 + 3) * LDAP + kk] = p11[1];
                sB3[(ii4 + 0) * LDAP + kk] = p30[0];
                sB3[(ii4 + 1) * LDAP + kk] = p30[1];
                sB3[(ii4 + 2) * LDAP + kk] = p31[0];
                sB3[(ii4 + 3) * LDAP + kk] = p31[1];
            }
        }
        __syncthreads();

        // ---- fragments + 8 WMMAs
        v16bf a[2];
        int ak = hiHalf ? 16 : 0;
        #pragma unroll
        for (int m = 0; m < 2; ++m) {
            int arow = wM * 32 + m * 16 + l15;
            #pragma unroll
            for (int p = 0; p < 8; ++p) {
                int kb = 2 * p + ak;
                bf16x2 tv = *(const bf16x2*)&sA[arow * LDAP + kb];
                a[m][2 * p] = tv[0];
                a[m][2 * p + 1] = tv[1];
            }
        }
        int bk = hiHalf ? 16 : 0;
        #pragma unroll
        for (int f = 0; f < 2; ++f) {
            int bcol = wN * 32 + f * 16 + l15;
            v16bf b1, b3;
            #pragma unroll
            for (int p = 0; p < 8; ++p) {
                bf16x2 t1 = *(const bf16x2*)&sB1[bcol * LDAP + 2 * p + bk];
                bf16x2 t3 = *(const bf16x2*)&sB3[bcol * LDAP + 2 * p + bk];
                b1[2 * p] = t1[0]; b1[2 * p + 1] = t1[1];
                b3[2 * p] = t3[0]; b3[2 * p + 1] = t3[1];
            }
            #pragma unroll
            for (int m = 0; m < 2; ++m) {
                accG[m][f] = wmma_bf16(a[m], b1, accG[m][f]);
                accU[m][f] = wmma_bf16(a[m], b3, accU[m][f]);
            }
        }
    }

    // ---- epilogue: fused silu(g)*u (fast exp + rcp), bf16 out
    #pragma unroll
    for (int m = 0; m < 2; ++m) {
        #pragma unroll
        for (int f = 0; f < 2; ++f) {
            int col = i0 + wN * 32 + f * 16 + l15;
            #pragma unroll
            for (int r = 0; r < 8; ++r) {
                int row = wM * 32 + m * 16 + r + (hiHalf ? 8 : 0);
                int pos = rowBase + row;
                if (pos < gEnd) {
                    float g = accG[m][f][r], u = accU[m][f][r];
                    float sig = __builtin_amdgcn_rcpf(1.f + __expf(-g));
                    hbuf[(size_t)pos * INTER + col] = (__bf16)(g * sig * u);
                }
            }
        }
    }
}

// =====================================================================
// 7) mlp2: y = h @ w2[e], scattered to combine slots (fp32 out)
//    grid: (16, HID/TN=32, NGROUPS=33), block 256
//    A tile is a pure bf16 copy -> async global->LDS when available
// =====================================================================
__global__ __launch_bounds__(256) void mlp2_kernel(
    const __bf16* __restrict__ hbuf,
    const float* __restrict__ w2, const float* __restrict__ sw2,
    const int* __restrict__ row_slot, const int* __restrict__ offsets,
    float* __restrict__ y)
{
    int e = blockIdx.z;
    int gStart = offsets[e], gEnd = offsets[e + 1];
    int rowBase = gStart + blockIdx.x * TM;
    if (rowBase >= gEnd) return;
    int i0 = blockIdx.y * TN;

    const float* B2 = (e < NUM_EXPERTS) ? (w2 + (size_t)e * INTER * HID) : sw2;

    __shared__ __bf16 sA[TM * LDAA];
    __shared__ __bf16 sB[TN * LDAP];
    __shared__ int sSlot[TM];

    int tid = threadIdx.x;
    int lane = tid & 31, wv = tid >> 5;
    int wM = wv & 3, wN = wv >> 2;
    int l15 = lane & 15;
    int hiHalf = (lane >= 16);

    if (tid < TM) {
        int p = rowBase + tid;
        sSlot[tid] = (p < gEnd) ? row_slot[p] : -1;
    }

    v8f acc[2][2] = {};

    for (int k0 = 0; k0 < INTER; k0 += TKK) {
        __syncthreads();
        // ---- stage A: 128 rows x 32 bf16 = 128 x 64B, 16B chunks per lane
        {
            int c  = tid & 3;          // 16B chunk within row
            int rb = tid >> 2;         // 0..63
            #pragma unroll
            for (int it = 0; it < 2; ++it) {
                int r = rb + it * 64;
                int p = rowBase + r;
                if (p >= gEnd) p = gStart;   // clamp; results discarded later
                const char* gsrc = (const char*)(hbuf + (size_t)p * INTER + k0) + c * 16;
                char* ldst = (char*)&sA[r * LDAA] + c * 16;
                async_copy16(gsrc, ldst);
            }
        }
        // ---- stage B transposed, float4 loads
        {
            int ii4 = (tid & 15) * 4;
            int kq  = tid >> 4;
            #pragma unroll
            for (int it = 0; it < 2; ++it) {
                int kk = kq + it * 16;
                size_t gi = (size_t)(k0 + kk) * HID + i0 + ii4;
                float4 a2 = *(const float4*)&B2[gi];
                if (k0 + TKK < INTER)
                    __builtin_prefetch(&B2[gi + (size_t)TKK * HID], 0, 1);
                bf16x2 p0 = f2bf2(a2.x, a2.y), p1 = f2bf2(a2.z, a2.w);
                sB[(ii4 + 0) * LDAP + kk] = p0[0];
                sB[(ii4 + 1) * LDAP + kk] = p0[1];
                sB[(ii4 + 2) * LDAP + kk] = p1[0];
                sB[(ii4 + 3) * LDAP + kk] = p1[1];
            }
        }
        wait_async();
        __syncthreads();

        v16bf a[2];
        int ak = hiHalf ? 16 : 0;
        #pragma unroll
        for (int m = 0; m < 2; ++m) {
            int arow = wM * 32 + m * 16 + l15;
            #pragma unroll
            for (int p = 0; p < 8; ++p) {
                int kb = 2 * p + ak;
                bf16x2 tv = *(const bf16x2*)&sA[arow * LDAA + kb];
                a[m][2 * p] = tv[0];
                a[m][2 * p + 1] = tv[1];
            }
        }
        int bk = hiHalf ? 16 : 0;
        #pragma unroll
        for (int f = 0; f < 2; ++f) {
            int bcol = wN * 32 + f * 16 + l15;
            v16bf b;
            #pragma unroll
            for (int p = 0; p < 8; ++p) {
                bf16x2 tv = *(const bf16x2*)&sB[bcol * LDAP + 2 * p + bk];
                b[2 * p] = tv[0];
                b[2 * p + 1] = tv[1];
            }
            #pragma unroll
            for (int m = 0; m < 2; ++m)
                acc[m][f] = wmma_bf16(a[m], b, acc[m][f]);
        }
    }

    #pragma unroll
    for (int m = 0; m < 2; ++m) {
        #pragma unroll
        for (int f = 0; f < 2; ++f) {
            int col = i0 + wN * 32 + f * 16 + l15;
            #pragma unroll
            for (int r = 0; r < 8; ++r) {
                int row = wM * 32 + m * 16 + r + (hiHalf ? 8 : 0);
                int slot = sSlot[row];
                if (slot >= 0)
                    y[(size_t)slot * HID + col] = acc[m][f][r];
            }
        }
    }
}

// =====================================================================
// 8) combine: out[t] = y_shared[t] + sum_k w[t,k] * y[t,k]   (fixed order)
// =====================================================================
__global__ __launch_bounds__(256) void combine_kernel(
    const float* __restrict__ y, const float* __restrict__ wts,
    float* __restrict__ out)
{
    int t = blockIdx.x;
    int h = blockIdx.y * 256 + threadIdx.x;
    float acc = y[(size_t)(NASSIGN + t) * HID + h];
    #pragma unroll
    for (int k = 0; k < TOPK; ++k)
        acc += wts[t * TOPK + k] * y[(size_t)(t * TOPK + k) * HID + h];
    out[(size_t)t * HID + h] = acc;
}

// =====================================================================
// host launcher
// =====================================================================
extern "C" void kernel_launch(void* const* d_in, const int* in_sizes, int n_in,
                              void* d_out, int out_size, void* d_ws, size_t ws_size,
                              hipStream_t stream) {
    const float* x    = (const float*)d_in[0];
    const float* gw   = (const float*)d_in[1];
    const float* gb   = (const float*)d_in[2];
    const float* w1   = (const float*)d_in[3];
    const float* w3   = (const float*)d_in[4];
    const float* w2   = (const float*)d_in[5];
    const float* sw1  = (const float*)d_in[6];
    const float* sw3  = (const float*)d_in[7];
    const float* sw2  = (const float*)d_in[8];
    float* out = (float*)d_out;

    char* ws = (char*)d_ws;
    size_t off = 0;
    auto alloc = [&](size_t bytes) -> void* {
        void* p = ws + off;
        off = (off + bytes + 255) & ~(size_t)255;
        return p;
    };
    int*    topk      = (int*)alloc(NASSIGN * sizeof(int));
    float*  wts       = (float*)alloc(NASSIGN * sizeof(float));
    int*    counts    = (int*)alloc(NUM_EXPERTS * sizeof(int));
    int*    offsets   = (int*)alloc((NGROUPS + 1) * sizeof(int));
    int*    cursor    = (int*)alloc(NUM_EXPERTS * sizeof(int));
    int*    row_token = (int*)alloc(NROWS * sizeof(int));
    int*    row_slot  = (int*)alloc(NROWS * sizeof(int));
    __bf16* hbuf      = (__bf16*)alloc((size_t)NROWS * INTER * sizeof(__bf16));
    float*  ybuf      = (float*)alloc((size_t)NROWS * HID * sizeof(float));
    (void)ws_size; (void)n_in; (void)in_sizes; (void)out_size;

    gate_kernel<<<NTOK, 32, 0, stream>>>(x, gw, gb, topk, wts);
    zero_counts_kernel<<<1, 64, 0, stream>>>(counts);
    hist_kernel<<<(NASSIGN + 255) / 256, 256, 0, stream>>>(topk, counts);
    scan_kernel<<<1, 32, 0, stream>>>(counts, offsets, cursor);
    scatter_kernel<<<(NROWS + 255) / 256, 256, 0, stream>>>(topk, cursor, row_token, row_slot);

    dim3 g1(NTOK / TM, INTER / TN, NGROUPS);  // (16, 22, 33)
    mlp1_kernel<<<g1, 256, 0, stream>>>(x, w1, w3, sw1, sw3, row_token, offsets, hbuf);

    dim3 g2(NTOK / TM, HID / TN, NGROUPS);    // (16, 32, 33)
    mlp2_kernel<<<g2, 256, 0, stream>>>(hbuf, w2, sw2, row_slot, offsets, ybuf);

    dim3 g3(NTOK, HID / 256);
    combine_kernel<<<g3, 256, 0, stream>>>(ybuf, wts, out);
}
